// ICTPBlock_71262097376094
// MI455X (gfx1250) — compile-verified, matrix operand order checked by reading, not run on hardware
//
#include <hip/hip_runtime.h>
#include <hip/hip_fp16.h>

// ---------------------------------------------------------------------------
// MI455X (gfx1250, wave32) equivariant message block.
// Per-edge radial MLPs (1->64->64->CH, 14 active keys) + 3 sgate MLPs run as
// f16 WMMA (v_wmma_f32_16x16x32_f16), 16 edges per wave tile.
// All WMMA B-fragments and the layer-1 W/b fragments are PRE-SHUFFLED into the
// per-lane register layout so each lane loads 2x global_load_b128 per fragment.
// Message accumulators (31MB) stay L2-resident; scatter via f32 global atomics.
// Xsrc gather uses GLOBAL_LOAD_ASYNC_TO_LDS_B128 (ASYNCcnt-tracked).
// ---------------------------------------------------------------------------

typedef __attribute__((ext_vector_type(16))) _Float16 v16h;
typedef __attribute__((ext_vector_type(8)))  _Float16 v8h;
typedef __attribute__((ext_vector_type(8)))  float    v8f;
typedef __attribute__((ext_vector_type(4)))  int      v4i;
typedef __attribute__((address_space(1))) v4i gas_v4i;   // global
typedef __attribute__((address_space(3))) v4i las_v4i;   // LDS

#if defined(__has_builtin)
#if __has_builtin(__builtin_amdgcn_global_load_async_to_lds_b128) && \
    __has_builtin(__builtin_amdgcn_s_wait_asynccnt)
#define HAVE_ASYNC_LDS 1
#endif
#endif
#ifndef HAVE_ASYNC_LDS
#define HAVE_ASYNC_LDS 0
#endif

#define NKEY 14
#define NSG  3
// radial key block layout (halfs) -- fragment-shuffled
#define OF_W1A0 0        // [32][16] layer1 W, K chunk 0, A-layout
#define OF_B1A0 512      // [32][16] layer1 b, K chunk 0
#define OF_W1A1 1024     // [32][16] layer1 W, K chunk 1
#define OF_B1A1 1536     // [32][16] layer1 b, K chunk 1
#define OF_W2F  2048     // [ntl*2+kc][32][16] layer2 B-fragments (8 frags)
#define OF_B2   6144     // [64]
#define OF_W3F  6208     // [ntl*2+kc][32][16] layer3 B-fragments (4 frags)
#define OF_B3   8256     // [32] padded
#define KEYSZP  8320     // halfs (byte size 16640, 32B multiple)
// sgate block layout (halfs)
#define SG_LNG  0
#define SG_LNB  32
#define SG_B1   64
#define SG_B2   96
#define SG_W1F  128      // [ntl][32][16] (2 frags)
#define SG_W2F  1152     // [ntl][32][16] (2 frags)
#define SGSZ    2176

__constant__ int dL1[NKEY] = {0,0,0, 1,1,1, 1,1,1, 1, 2,2,2, 3};
__constant__ int dL3[NKEY] = {1,2,3, 0,1,2, 1,2,3, 2, 1,2,3, 2};
__constant__ int dCH[4] = {32,16,8,4};
__constant__ int dNC[4] = {1,3,9,27};

struct WPtrs {
    const float* rad[NKEY][6];   // W1,b1,W2,b2,W3,b3
    const float* sg[NSG][6];     // ln_g, ln_b, W1, b1, W2, b2
};

__device__ __forceinline__ float sigmf(float x) {
    return __builtin_amdgcn_rcpf(1.0f + __expf(-x));     // v_rcp_f32, no IEEE div
}
__device__ __forceinline__ float siluf(float x) { return x * sigmf(x); }
__device__ __forceinline__ float tanhfast(float x) { return 2.0f * sigmf(2.0f * x) - 1.0f; }

__device__ __forceinline__ v16h cat8(v8h lo, v8h hi) {
    return __builtin_shufflevector(lo, hi, 0,1,2,3,4,5,6,7,8,9,10,11,12,13,14,15);
}
// fragment load: 32B contiguous per lane -> 2x b128
__device__ __forceinline__ v16h ldfrag_g(const _Float16* base) {
    const v8h* p = (const v8h*)base;
    return cat8(p[0], p[1]);
}
__device__ __forceinline__ v16h ldfrag_l(const _Float16* r0, const _Float16* r1) {
    return cat8(*(const v8h*)r0, *(const v8h*)r1);       // 2x ds_load_b128
}

__device__ __forceinline__ void st2p(const float* T, float* O) {
    float tr = (T[0] + T[4] + T[8]) * (1.0f / 3.0f);
#pragma unroll
    for (int i = 0; i < 9; ++i) O[i] = T[i];
    O[0] -= tr; O[4] -= tr; O[8] -= tr;
}

// Literal transcription of reference st_rank3 (note its asymmetric trace term).
__device__ __forceinline__ void st3p(const float* T, float* O) {
    float Ts[27];
#pragma unroll
    for (int a = 0; a < 3; ++a)
#pragma unroll
        for (int b = 0; b < 3; ++b)
#pragma unroll
            for (int c = 0; c < 3; ++c)
                Ts[a*9+b*3+c] = (T[a*9+b*3+c] + T[a*9+c*3+b] + T[b*9+a*3+c] +
                                 T[b*9+c*3+a] + T[c*9+a*3+b] + T[c*9+b*3+a]) * (1.0f/6.0f);
    float cv[3];
#pragma unroll
    for (int i = 0; i < 3; ++i) cv[i] = Ts[i*9+0] + Ts[i*9+4] + Ts[i*9+8];
#pragma unroll
    for (int x = 0; x < 3; ++x)
#pragma unroll
        for (int y = 0; y < 3; ++y)
#pragma unroll
            for (int z = 0; z < 3; ++z)
                O[x*9+y*3+z] = Ts[x*9+y*3+z]
                             - ((y == z ? 2.0f * cv[x] : 0.0f) + (x == y ? cv[z] : 0.0f)) * 0.2f;
}

// ---------------------------------------------------------------------------
// Kernel 1: repack weights to f16, pre-shuffled into WMMA fragment layout.
// A/B 16-bit fragment mapping (CDNA5 ISA): lane L, half t (0..15):
//   i=t>>1, p=t&1, k = ((i>>2)<<4) + ((L>>4)<<3) + ((i&3)<<1) + p, n=(L&15)
// ---------------------------------------------------------------------------
__global__ void prep_weights(WPtrs P, _Float16* __restrict__ radW, _Float16* __restrict__ sgW) {
    int sec = blockIdx.x, t = threadIdx.x, nt = blockDim.x;
    if (sec < NKEY) {
        _Float16* D = radW + (size_t)sec * KEYSZP;
        const float *W1 = P.rad[sec][0], *b1 = P.rad[sec][1], *W2 = P.rad[sec][2],
                    *b2 = P.rad[sec][3], *W3 = P.rad[sec][4], *b3 = P.rad[sec][5];
        int ch = dCH[dL3[sec]];
        for (int idx = t; idx < 512; idx += nt) {           // layer-1 fragments
            int L = idx >> 4, tt = idx & 15;
            int i = tt >> 1, p = tt & 1;
            int k = ((i >> 2) << 4) + ((L >> 4) << 3) + ((i & 3) << 1) + p;
            D[OF_W1A0 + idx] = (_Float16)W1[k];
            D[OF_B1A0 + idx] = (_Float16)b1[k];
            D[OF_W1A1 + idx] = (_Float16)W1[k + 32];
            D[OF_B1A1 + idx] = (_Float16)b1[k + 32];
        }
        for (int idx = t; idx < 4096; idx += nt) {          // layer-2 B fragments
            int fr = idx >> 9, rem = idx & 511;
            int L = rem >> 4, tt = rem & 15;
            int i = tt >> 1, p = tt & 1;
            int ntl = fr >> 1, kc = fr & 1;
            int n = (ntl << 4) + (L & 15);
            int k = (kc << 5) + ((i >> 2) << 4) + ((L >> 4) << 3) + ((i & 3) << 1) + p;
            D[OF_W2F + idx] = (_Float16)W2[k * 64 + n];
        }
        for (int idx = t; idx < 64; idx += nt) D[OF_B2 + idx] = (_Float16)b2[idx];
        for (int idx = t; idx < 2048; idx += nt) {          // layer-3 B fragments
            int fr = idx >> 9, rem = idx & 511;
            int L = rem >> 4, tt = rem & 15;
            int i = tt >> 1, p = tt & 1;
            int ntl = fr >> 1, kc = fr & 1;
            int n = (ntl << 4) + (L & 15);
            int k = (kc << 5) + ((i >> 2) << 4) + ((L >> 4) << 3) + ((i & 3) << 1) + p;
            D[OF_W3F + idx] = (_Float16)((n < ch) ? W3[k * ch + n] : 0.0f);
        }
        for (int idx = t; idx < 32; idx += nt) D[OF_B3 + idx] = (_Float16)((idx < ch) ? b3[idx] : 0.0f);
    } else {
        int s = sec - NKEY;
        _Float16* D = sgW + (size_t)s * SGSZ;
        const float *lng = P.sg[s][0], *lnb = P.sg[s][1], *W1 = P.sg[s][2],
                    *b1 = P.sg[s][3], *W2 = P.sg[s][4], *b2 = P.sg[s][5];
        int ch = dCH[s + 1];
        for (int idx = t; idx < 32; idx += nt) {
            D[SG_LNG + idx] = (_Float16)lng[idx];
            D[SG_LNB + idx] = (_Float16)lnb[idx];
            D[SG_B1 + idx]  = (_Float16)b1[idx];
            D[SG_B2 + idx]  = (_Float16)((idx < ch) ? b2[idx] : 0.0f);
        }
        for (int idx = t; idx < 1024; idx += nt) {          // 2 frags each
            int fr = idx >> 9, rem = idx & 511;
            int L = rem >> 4, tt = rem & 15;
            int i = tt >> 1, p = tt & 1;
            int n = (fr << 4) + (L & 15);
            int k = ((i >> 2) << 4) + ((L >> 4) << 3) + ((i & 3) << 1) + p;
            D[SG_W1F + idx] = (_Float16)W1[k * 32 + n];
            D[SG_W2F + idx] = (_Float16)((n < ch) ? W2[k * ch + n] : 0.0f);
        }
    }
}

// ---------------------------------------------------------------------------
// Kernel 2: fused per-edge kernel. 1 wave32 per 16-edge tile.
// ---------------------------------------------------------------------------
__global__ void __launch_bounds__(32) edge_kernel(
    const float* __restrict__ pos, const int* __restrict__ ei,
    const float* __restrict__ f0, const float* __restrict__ f1,
    const float* __restrict__ f2, const float* __restrict__ f3,
    const _Float16* __restrict__ radW, const _Float16* __restrict__ sgW,
    float* __restrict__ msg0, float* __restrict__ msg1,
    float* __restrict__ msg2, float* __restrict__ msg3, int E)
{
    __shared__ float sU1[16][3];
    __shared__ float sR[16];
    __shared__ float sU2[16][9];
    __shared__ float sU3[16][27];
    __shared__ float sM1[16][3];
    __shared__ float sM2[16][9];
    __shared__ float sM3[16][27];
    __shared__ __align__(16) float sX0[16][32];
    __shared__ int   sDst[16];
    __shared__ int   sSrc[16];
    __shared__ __align__(16) _Float16 sH[16][64];
    __shared__ float sAlpha[16][32];
    __shared__ float sBase[16][27];

    const int lane = threadIdx.x;
    const int m    = lane & 15;
    const int hlf  = lane >> 4;

    if (lane < 16) {
        int e = blockIdx.x * 16 + lane;
        bool valid = e < E;
        int ec = valid ? e : (E - 1);
        sDst[lane] = valid ? ei[ec] : -1;
        sSrc[lane] = ei[E + ec];
    }
    __syncthreads();

#if HAVE_ASYNC_LDS
    // Xsrc gather: 16 rows x 128B = 128 chunks of 16B; async copy to LDS.
    // One instruction moves 32 lanes x 16B, tracked by ASYNCcnt.
#pragma unroll
    for (int it = 0; it < 4; ++it) {
        int cidx = lane + 32 * it;
        int row = cidx >> 3, j = cidx & 7;
        const float* gp = f0 + (size_t)sSrc[row] * 32 + j * 4;
        __builtin_amdgcn_global_load_async_to_lds_b128(
            (gas_v4i*)gp, (las_v4i*)&sX0[row][j * 4], 0, 0);
    }
#endif

    if (lane < 16) {
        int nd = sDst[lane] >= 0 ? sDst[lane] : ei[E - 1];
        int ns = sSrc[lane];
        float dx = pos[nd*3+0] - pos[ns*3+0];
        float dy = pos[nd*3+1] - pos[ns*3+1];
        float dz = pos[nd*3+2] - pos[ns*3+2];
        float rn = sqrtf(dx*dx + dy*dy + dz*dz) + 1e-9f;
        float uu[3] = {dx/rn, dy/rn, dz/rn};
        sR[lane] = rn;
        sU1[lane][0] = uu[0]; sU1[lane][1] = uu[1]; sU1[lane][2] = uu[2];
        float tr2 = uu[0]*uu[0] + uu[1]*uu[1] + uu[2]*uu[2];
#pragma unroll
        for (int i = 0; i < 3; ++i)
#pragma unroll
            for (int j = 0; j < 3; ++j)
                sU2[lane][i*3+j] = uu[i]*uu[j] - (i == j ? tr2 * (1.0f/3.0f) : 0.0f);
        {
            float T[27], O[27];
#pragma unroll
            for (int i = 0; i < 3; ++i)
#pragma unroll
                for (int j = 0; j < 3; ++j)
#pragma unroll
                    for (int k = 0; k < 3; ++k) T[i*9+j*3+k] = uu[i]*uu[j]*uu[k];
            st3p(T, O);
            for (int q = 0; q < 27; ++q) sU3[lane][q] = O[q];
        }
        // channel means (contiguous loads per channel, reduce in registers)
        {
            float a[3] = {0,0,0};
            const float* p1 = f1 + (size_t)ns * 48;
            for (int c = 0; c < 16; ++c)
                for (int q = 0; q < 3; ++q) a[q] += p1[c*3+q];
            for (int q = 0; q < 3; ++q) sM1[lane][q] = a[q] * (1.0f/16.0f);
        }
        {
            float a[9];
            for (int q = 0; q < 9; ++q) a[q] = 0.0f;
            const float* p2 = f2 + (size_t)ns * 72;
            for (int c = 0; c < 8; ++c)
                for (int q = 0; q < 9; ++q) a[q] += p2[c*9+q];
            for (int q = 0; q < 9; ++q) sM2[lane][q] = a[q] * 0.125f;
        }
        {
            float a[27];
            for (int q = 0; q < 27; ++q) a[q] = 0.0f;
            const float* p3 = f3 + (size_t)ns * 108;
            for (int c = 0; c < 4; ++c)
                for (int q = 0; q < 27; ++q) a[q] += p3[c*27+q];
            for (int q = 0; q < 27; ++q) sM3[lane][q] = a[q] * 0.25f;
        }
#if !HAVE_ASYNC_LDS
        const float* p0 = f0 + (size_t)ns * 32;
        for (int j = 0; j < 32; ++j) sX0[lane][j] = p0[j];
#endif
    }
#if HAVE_ASYNC_LDS
    __builtin_amdgcn_s_wait_asynccnt(0);
#endif
    __syncthreads();

    for (int kk = 0; kk < NKEY; ++kk) {
        const _Float16* W = radW + (size_t)kk * KEYSZP;
        if (kk + 1 < NKEY)
            __builtin_prefetch(radW + (size_t)(kk + 1) * KEYSZP, 0, 1);  // global_prefetch_b8
        const int l3 = dL3[kk];
        const int ch = dCH[l3];
        const int nc = dNC[l3];
        const int lane16 = lane * 16;

        // ---- layer 1 (1 -> 64): packed f16 FMA straight into A fragments
        _Float16 rmh = (_Float16)sR[m];
        v16h x0 = ldfrag_g(W + OF_W1A0 + lane16) * rmh + ldfrag_g(W + OF_B1A0 + lane16);
        v16h x1 = ldfrag_g(W + OF_W1A1 + lane16) * rmh + ldfrag_g(W + OF_B1A1 + lane16);
        v16h a0, a1;
#pragma unroll
        for (int tt = 0; tt < 16; ++tt) {
            a0[tt] = (_Float16)siluf((float)x0[tt]);
            a1[tt] = (_Float16)siluf((float)x1[tt]);
        }

        // ---- layer 2 (64 -> 64): 4 N-tiles x 2 K-chunks
#pragma unroll
        for (int ntl = 0; ntl < 4; ++ntl) {
            int n = (ntl << 4) + m;
            v8f c = {};
            c = __builtin_amdgcn_wmma_f32_16x16x32_f16(false, a0, false,
                    ldfrag_g(W + OF_W2F + ((ntl << 1) + 0) * 512 + lane16), (short)0, c, false, false);
            c = __builtin_amdgcn_wmma_f32_16x16x32_f16(false, a1, false,
                    ldfrag_g(W + OF_W2F + ((ntl << 1) + 1) * 512 + lane16), (short)0, c, false, false);
            float bias = (float)W[OF_B2 + n];
#pragma unroll
            for (int v = 0; v < 8; ++v) sH[(hlf << 3) + v][n] = (_Float16)siluf(c[v] + bias);
        }
        __syncthreads();

        // ---- layer 3 (64 -> 32 padded): A frags = contiguous 16B runs of sH
        v16h h0 = ldfrag_l(&sH[m][hlf * 8],      &sH[m][16 + hlf * 8]);
        v16h h1 = ldfrag_l(&sH[m][32 + hlf * 8], &sH[m][48 + hlf * 8]);
#pragma unroll
        for (int ntl = 0; ntl < 2; ++ntl) {
            int n = (ntl << 4) + m;
            v8f c = {};
            c = __builtin_amdgcn_wmma_f32_16x16x32_f16(false, h0, false,
                    ldfrag_g(W + OF_W3F + ((ntl << 1) + 0) * 512 + lane16), (short)0, c, false, false);
            c = __builtin_amdgcn_wmma_f32_16x16x32_f16(false, h1, false,
                    ldfrag_g(W + OF_W3F + ((ntl << 1) + 1) * 512 + lane16), (short)0, c, false, false);
            float bias = (float)W[OF_B3 + n];
#pragma unroll
            for (int v = 0; v < 8; ++v) sAlpha[(hlf << 3) + v][n] = c[v] + bias;
        }
        __syncthreads();

        // ---- sgate path for l1==0 keys (kk < 3), fully WMMA (K=32)
        if (dL1[kk] == 0) {
            const _Float16* S = sgW + (size_t)kk * SGSZ;
            if (lane < 16) {
                float mean = 0.0f;
                for (int j = 0; j < 32; ++j) mean += sX0[lane][j];
                mean *= (1.0f/32.0f);
                float var = 0.0f;
                for (int j = 0; j < 32; ++j) { float d = sX0[lane][j] - mean; var += d*d; }
                var *= (1.0f/32.0f);
                float inv = rsqrtf(var + 1e-5f);
                for (int j = 0; j < 32; ++j)
                    sH[lane][j] = (_Float16)((sX0[lane][j] - mean) * inv * (float)S[SG_LNG+j]
                                             + (float)S[SG_LNB+j]);
            }
            __syncthreads();
            v16h xa = ldfrag_l(&sH[m][hlf * 8], &sH[m][16 + hlf * 8]);
#pragma unroll
            for (int ntl = 0; ntl < 2; ++ntl) {
                int n = (ntl << 4) + m;
                v8f c = {};
                c = __builtin_amdgcn_wmma_f32_16x16x32_f16(false, xa, false,
                        ldfrag_g(S + SG_W1F + ntl * 512 + lane16), (short)0, c, false, false);
                float bias = (float)S[SG_B1 + n];
#pragma unroll
                for (int v = 0; v < 8; ++v) sH[(hlf << 3) + v][32 + n] = (_Float16)siluf(c[v] + bias);
            }
            __syncthreads();
            v16h ha = ldfrag_l(&sH[m][32 + hlf * 8], &sH[m][48 + hlf * 8]);
#pragma unroll
            for (int ntl = 0; ntl < 2; ++ntl) {
                int n = (ntl << 4) + m;
                v8f c = {};
                c = __builtin_amdgcn_wmma_f32_16x16x32_f16(false, ha, false,
                        ldfrag_g(S + SG_W2F + ntl * 512 + lane16), (short)0, c, false, false);
                float bias = (float)S[SG_B2 + n];
#pragma unroll
                for (int v = 0; v < 8; ++v)
                    sAlpha[(hlf << 3) + v][n] += 0.1f * tanhfast(c[v] + bias);
            }
            __syncthreads();
        }

        // ---- tensor bases per edge (lanes 0..15)
        if (lane < 16) {
            float* B = sBase[lane];
            const float* u  = sU1[lane];
            const float* m1 = sM1[lane];
            const float* m2 = sM2[lane];
            const float* m3 = sM3[lane];
            switch (kk) {
            case 0: B[0] = u[0]; B[1] = u[1]; B[2] = u[2]; break;                        // (0,1,1)
            case 1: for (int q = 0; q < 9; ++q)  B[q] = sU2[lane][q]; break;             // (0,2,2)
            case 2: for (int q = 0; q < 27; ++q) B[q] = sU3[lane][q]; break;             // (0,3,3)
            case 3: B[0] = m1[0]*u[0] + m1[1]*u[1] + m1[2]*u[2]; break;                  // (1,1,0)
            case 4:                                                                       // (1,1,1)
                B[0] = m1[1]*u[2] - m1[2]*u[1];
                B[1] = m1[2]*u[0] - m1[0]*u[2];
                B[2] = m1[0]*u[1] - m1[1]*u[0];
                break;
            case 5: {                                                                     // (1,1,2)
                float T[9];
                for (int i = 0; i < 3; ++i)
                    for (int j = 0; j < 3; ++j) T[i*3+j] = 0.5f*(m1[i]*u[j] + m1[j]*u[i]);
                st2p(T, B);
            } break;
            case 6:                                                                       // (1,2,1)
                for (int j = 0; j < 3; ++j)
                    B[j] = sU2[lane][j*3+0]*m1[0] + sU2[lane][j*3+1]*m1[1] + sU2[lane][j*3+2]*m1[2];
                break;
            case 7: {                                                                     // (1,2,2)
                float v[3], T[9];
                for (int j = 0; j < 3; ++j)
                    v[j] = sU2[lane][j*3+0]*m1[0] + sU2[lane][j*3+1]*m1[1] + sU2[lane][j*3+2]*m1[2];
                for (int i = 0; i < 3; ++i)
                    for (int j = 0; j < 3; ++j) T[i*3+j] = 0.5f*(m1[i]*v[j] + m1[j]*v[i]);
                st2p(T, B);
            } break;
            case 8: {                                                                     // (1,2,3)
                float T[27];
                for (int i = 0; i < 3; ++i)
                    for (int j = 0; j < 3; ++j)
                        for (int k = 0; k < 3; ++k) T[i*9+j*3+k] = m1[i]*sU2[lane][j*3+k];
                st3p(T, B);
            } break;
            case 9: {                                                                     // (1,3,2)
                float c3[3], T[9];
                for (int j = 0; j < 3; ++j)
                    c3[j] = sU3[lane][j*9+0] + sU3[lane][j*9+4] + sU3[lane][j*9+8];
                for (int i = 0; i < 3; ++i)
                    for (int j = 0; j < 3; ++j) T[i*3+j] = m1[i]*c3[j];
                st2p(T, B);
            } break;
            case 10:                                                                      // (2,1,1)
                for (int j = 0; j < 3; ++j)
                    B[j] = m2[j*3+0]*u[0] + m2[j*3+1]*u[1] + m2[j*3+2]*u[2];
                break;
            case 11: {                                                                    // (2,1,2)
                float v[3], T[9];
                for (int j = 0; j < 3; ++j)
                    v[j] = m2[j*3+0]*u[0] + m2[j*3+1]*u[1] + m2[j*3+2]*u[2];
                for (int i = 0; i < 3; ++i)
                    for (int j = 0; j < 3; ++j) T[i*3+j] = 0.5f*(u[i]*v[j] + u[j]*v[i]);
                st2p(T, B);
            } break;
            case 12: {                                                                    // (2,1,3)
                float T[27];
                for (int i = 0; i < 3; ++i)
                    for (int j = 0; j < 3; ++j)
                        for (int k = 0; k < 3; ++k) T[i*9+j*3+k] = u[i]*m2[j*3+k];
                st3p(T, B);
            } break;
            default: {                                                                    // (3,1,2)
                float c3[3], T[9];
                for (int j = 0; j < 3; ++j)
                    c3[j] = m3[j*9+0] + m3[j*9+4] + m3[j*9+8];
                for (int i = 0; i < 3; ++i)
                    for (int j = 0; j < 3; ++j) T[i*3+j] = u[i]*c3[j];
                st2p(T, B);
            } break;
            }
        }
        __syncthreads();

        // ---- scatter-add into L2-resident message accumulators
        float* msgp = (l3 == 0) ? msg0 : (l3 == 1) ? msg1 : (l3 == 2) ? msg2 : msg3;
        const int per = ch * nc;
        const int tot = 16 * per;
        for (int idx = lane; idx < tot; idx += 32) {
            int e = idx / per;
            int rem = idx - e * per;
            int c = rem / nc;
            int comp = rem - c * nc;
            int d = sDst[e];
            if (d >= 0)
                atomicAdd(msgp + (size_t)d * per + (size_t)c * nc + comp,
                          sAlpha[e][c] * sBase[e][comp]);
        }
        __syncthreads();
    }
}

// ---------------------------------------------------------------------------
// Kernel 3: per-node gate MLP + residual + traceless projections.
// ---------------------------------------------------------------------------
__global__ void node_kernel(
    const float* __restrict__ f0, const float* __restrict__ f1,
    const float* __restrict__ f2, const float* __restrict__ f3,
    const float* __restrict__ gW1, const float* __restrict__ gb1,
    const float* __restrict__ gW2, const float* __restrict__ gb2,
    const float* __restrict__ msg0, const float* __restrict__ msg1,
    const float* __restrict__ msg2, const float* __restrict__ msg3,
    float* __restrict__ out, int N)
{
    int n = blockIdx.x * blockDim.x + threadIdx.x;
    if (n >= N) return;

    float iv[4];
    {
        float s = 0.0f; const float* p = f0 + (size_t)n*32;
        for (int i = 0; i < 32; ++i) s += p[i]*p[i];
        iv[0] = sqrtf(s * (1.0f/32.0f) + 1e-12f);
    }
    {
        float s = 0.0f; const float* p = f1 + (size_t)n*48;
        for (int i = 0; i < 48; ++i) s += p[i]*p[i];
        iv[1] = sqrtf(s * (1.0f/48.0f) + 1e-12f);
    }
    {
        float s = 0.0f; const float* p = f2 + (size_t)n*72;
        for (int i = 0; i < 72; ++i) s += p[i]*p[i];
        iv[2] = sqrtf(s * (1.0f/72.0f) + 1e-12f);
    }
    {
        float s = 0.0f; const float* p = f3 + (size_t)n*108;
        for (int i = 0; i < 108; ++i) s += p[i]*p[i];
        iv[3] = sqrtf(s * (1.0f/108.0f) + 1e-12f);
    }

    float h[120];
    for (int j = 0; j < 120; ++j) {
        float s = gb1[j];
        for (int i = 0; i < 60; ++i) {
            float vi = (i < 32) ? iv[0] : (i < 48) ? iv[1] : (i < 56) ? iv[2] : iv[3];
            s += vi * gW1[i*120 + j];
        }
        h[j] = siluf(s);
    }
    float g[60];
    for (int j = 0; j < 60; ++j) {
        float s = gb2[j];
        for (int i = 0; i < 120; ++i) s += h[i] * gW2[i*60 + j];
        g[j] = sigmf(s);
    }

    float* o0 = out;
    float* o1 = out + (size_t)N * 32;
    float* o2 = o1  + (size_t)N * 48;
    float* o3 = o2  + (size_t)N * 72;

    for (int c = 0; c < 32; ++c)
        o0[(size_t)n*32 + c] = f0[(size_t)n*32 + c] + msg0[(size_t)n*32 + c] * g[c];
    for (int c = 0; c < 16; ++c)
        for (int q = 0; q < 3; ++q) {
            size_t off = (size_t)n*48 + c*3 + q;
            o1[off] = f1[off] + msg1[off] * g[32 + c];
        }
    for (int c = 0; c < 8; ++c) {
        float T[9], O[9];
        for (int q = 0; q < 9; ++q) {
            size_t off = (size_t)n*72 + c*9 + q;
            T[q] = f2[off] + msg2[off] * g[48 + c];
        }
        st2p(T, O);
        for (int q = 0; q < 9; ++q) o2[(size_t)n*72 + c*9 + q] = O[q];
    }
    for (int c = 0; c < 4; ++c) {
        float T[27], O[27];
        for (int q = 0; q < 27; ++q) {
            size_t off = (size_t)n*108 + c*27 + q;
            T[q] = f3[off] + msg3[off] * g[56 + c];
        }
        st3p(T, O);
        for (int q = 0; q < 27; ++q) o3[(size_t)n*108 + c*27 + q] = O[q];
    }
}

// ---------------------------------------------------------------------------
// Host launch
// ---------------------------------------------------------------------------
extern "C" void kernel_launch(void* const* d_in, const int* in_sizes, int n_in,
                              void* d_out, int out_size, void* d_ws, size_t ws_size,
                              hipStream_t stream) {
    (void)n_in; (void)out_size; (void)ws_size;

    const float* pos = (const float*)d_in[0];
    const int*   ei  = (const int*)d_in[1];
    const float* f0  = (const float*)d_in[2];
    const float* f1  = (const float*)d_in[3];
    const float* f2  = (const float*)d_in[4];
    const float* f3  = (const float*)d_in[5];
    const int N = in_sizes[0] / 3;
    const int E = in_sizes[1] / 2;

    // Enumerate rad-dict keys in reference insertion order to find d_in bases.
    int allL[40][3], allBase[40], cnt = 0, idx = 6;
    for (int l1 = 0; l1 < 4; ++l1)
        for (int l2 = 0; l2 < 4; ++l2) {
            int lo = (l1 > l2) ? (l1 - l2) : (l2 - l1);
            int hi = (l1 + l2 < 3) ? (l1 + l2) : 3;
            for (int l3 = lo; l3 <= hi; ++l3) {
                allL[cnt][0] = l1; allL[cnt][1] = l2; allL[cnt][2] = l3;
                allBase[cnt] = idx; idx += 6; ++cnt;
            }
        }
    int sgBase[3];
    for (int s = 0; s < 3; ++s) { sgBase[s] = idx; idx += 6; }
    const float* gW1 = (const float*)d_in[idx + 0];
    const float* gb1 = (const float*)d_in[idx + 1];
    const float* gW2 = (const float*)d_in[idx + 2];
    const float* gb2 = (const float*)d_in[idx + 3];

    static const int AK[NKEY][3] = {
        {0,1,1},{0,2,2},{0,3,3},{1,1,0},{1,1,1},{1,1,2},{1,2,1},
        {1,2,2},{1,2,3},{1,3,2},{2,1,1},{2,1,2},{2,1,3},{3,1,2}};
    WPtrs P;
    for (int k = 0; k < NKEY; ++k)
        for (int j = 0; j < cnt; ++j)
            if (allL[j][0] == AK[k][0] && allL[j][1] == AK[k][1] && allL[j][2] == AK[k][2])
                for (int t = 0; t < 6; ++t)
                    P.rad[k][t] = (const float*)d_in[allBase[j] + t];
    for (int s = 0; s < NSG; ++s)
        for (int t = 0; t < 6; ++t)
            P.sg[s][t] = (const float*)d_in[sgBase[s] + t];

    // Workspace layout: f16 weight blocks | f32 message accumulators.
    char* ws = (char*)d_ws;
    _Float16* radW = (_Float16*)ws;
    _Float16* sgW  = (_Float16*)(ws + (size_t)NKEY * KEYSZP * 2);
    size_t msgOff = ((size_t)NKEY * KEYSZP + (size_t)NSG * SGSZ) * 2;
    msgOff = (msgOff + 255) & ~(size_t)255;
    float* msg0 = (float*)(ws + msgOff);
    float* msg1 = msg0 + (size_t)N * 32;
    float* msg2 = msg1 + (size_t)N * 48;
    float* msg3 = msg2 + (size_t)N * 72;
    size_t msgBytes = (size_t)N * (32 + 48 + 72 + 108) * sizeof(float);

    (void)hipMemsetAsync(ws + msgOff, 0, msgBytes, stream);
    prep_weights<<<NKEY + NSG, 256, 0, stream>>>(P, radW, sgW);
    edge_kernel<<<(E + 15) / 16, 32, 0, stream>>>(pos, ei, f0, f1, f2, f3,
                                                  radW, sgW, msg0, msg1, msg2, msg3, E);
    node_kernel<<<(N + 127) / 128, 128, 0, stream>>>(f0, f1, f2, f3, gW1, gb1, gW2, gb2,
                                                     msg0, msg1, msg2, msg3, (float*)d_out, N);
}
